// ModelSquad_38525856645633
// MI455X (gfx1250) — compile-verified, matrix operand order checked by reading, not run on hardware
//
#include <hip/hip_runtime.h>

// DMN SQuAD model for MI455X (gfx1250): all matmuls via v_wmma_f32_16x16x32_bf16
// (f32 operands converted to bf16 on the fly, f32 accumulate), LDS-staged tiles
// shared by 4 waves per workgroup.
// B=64, T=512, Tq=32, H=512, V=50000, HOPS=3.

namespace {

constexpr int Bc  = 64;
constexpr int Tc  = 512;
constexpr int TQc = 32;
constexpr int Hc  = 512;

typedef __attribute__((ext_vector_type(16))) __bf16 v16bf;
typedef __attribute__((ext_vector_type(8)))  __bf16 v8bf;
typedef __attribute__((ext_vector_type(8)))  float  v8f;

__device__ __forceinline__ float sigmf(float x) { return 1.0f / (1.0f + __expf(-x)); }

__device__ __forceinline__ float applyact(float x, int act) {
  switch (act) {
    case 1: return fmaxf(x, 0.0f);
    case 2: return tanhf(x);
    case 3: return sigmf(x);
    default: return x;
  }
}

__device__ __forceinline__ v16bf cat8(v8bf lo, v8bf hi) {
  return __builtin_shufflevector(lo, hi, 0, 1, 2, 3, 4, 5, 6, 7,
                                 8, 9, 10, 11, 12, 13, 14, 15);
}

// -------- generic GEMM: C[M,N] = act(A[M,K] @ B[K,N] + bias + Cin) -----------
// 4 waves / block, 32x32 macro-tile, each wave one 16x16 WMMA tile.
// K multiple of 32; M,N multiples of 32. Cooperative float4 (b128) global
// loads -> bf16 convert -> LDS stage (16B-aligned rows, stride 40) -> WMMA.
// Lane layout per CDNA5 ISA 7.12.2:
//   A (bf16 16x32): lane m = lane&15, kbase = (lane>>4)*8;
//                   a[i] <-> k = kbase+i, a[8+i] <-> k = kbase+16+i
//   B (bf16 32x16): lane n = lane&15, same k striping per VGPR.
//   C (f32 16x16):  lane n = lane&15, acc[v] <-> m = v + 8*(lane>>4).
__global__ __launch_bounds__(128) void gemm_wmma_bf16(
    const float* __restrict__ A, long lda,
    const float* __restrict__ Bm, long ldb,
    const float* __restrict__ Cin, long ldcin,
    const float* __restrict__ bias,
    float* __restrict__ C, long ldc,
    int M, int N, int K, int act)
{
  __shared__ __align__(16) __bf16 lA[32][40];
  __shared__ __align__(16) __bf16 lB[32][40];

  const int tid  = threadIdx.x;
  const int wid  = tid >> 5;
  const int lane = tid & 31;
  const int half = lane >> 4;
  const int idx  = lane & 15;
  const int n0 = blockIdx.x * 32;
  const int m0 = blockIdx.y * 32;
  const int mi = wid >> 1, ni = wid & 1;   // 2x2 sub-tiles of 16x16

  const int lr  = tid >> 2;        // 0..31: tile row for cooperative load
  const int lc8 = (tid & 3) * 8;   // 0,8,16,24: 8-wide column chunk

  const int kb = half * 8;
  const int nn = ni * 16 + idx;

  v8f acc = {};
  for (int k0 = 0; k0 < K; k0 += 32) {
    // stage A tile (rows m0+lr, k = k0+lc8 .. +7)
    {
      const float* ap = A + (long)(m0 + lr) * lda + k0 + lc8;
      const float4 a0 = *(const float4*)ap;
      const float4 a1 = *(const float4*)(ap + 4);
      v8bf av;
      av[0] = (__bf16)a0.x; av[1] = (__bf16)a0.y;
      av[2] = (__bf16)a0.z; av[3] = (__bf16)a0.w;
      av[4] = (__bf16)a1.x; av[5] = (__bf16)a1.y;
      av[6] = (__bf16)a1.z; av[7] = (__bf16)a1.w;
      *(v8bf*)&lA[lr][lc8] = av;
    }
    // stage B tile (k = k0+lr, n = n0+lc8 .. +7)
    {
      const float* bp = Bm + (long)(k0 + lr) * ldb + n0 + lc8;
      const float4 b0 = *(const float4*)bp;
      const float4 b1 = *(const float4*)(bp + 4);
      v8bf bv;
      bv[0] = (__bf16)b0.x; bv[1] = (__bf16)b0.y;
      bv[2] = (__bf16)b0.z; bv[3] = (__bf16)b0.w;
      bv[4] = (__bf16)b1.x; bv[5] = (__bf16)b1.y;
      bv[6] = (__bf16)b1.z; bv[7] = (__bf16)b1.w;
      *(v8bf*)&lB[lr][lc8] = bv;
    }
    __syncthreads();

    const int ar = mi * 16 + idx;
    const v16bf a = cat8(*(const v8bf*)&lA[ar][kb],
                         *(const v8bf*)&lA[ar][kb + 16]);
    v16bf b;
#pragma unroll
    for (int i = 0; i < 8; ++i) {
      b[i]     = lB[kb + i][nn];
      b[i + 8] = lB[kb + 16 + i][nn];
    }
    acc = __builtin_amdgcn_wmma_f32_16x16x32_bf16(false, a, false, b,
                                                  (short)0, acc, false, false);
    __syncthreads();
  }

  const int tn = n0 + nn;
  const int tm = m0 + mi * 16;
  const float bv = bias ? bias[tn] : 0.0f;
#pragma unroll
  for (int v = 0; v < 8; ++v) {
    const int m = tm + half * 8 + v;
    float x = acc[v] + bv;
    if (Cin) x += Cin[(long)m * ldcin + tn];
    C[(long)m * ldc + tn] = applyact(x, act);
  }
}

// -------- attention hidden layer: tanh(feats[B*T,7H] @ W1[7H,H] + b1) --------
// feats built on the fly: [f, M, q, f*q, f*M, |f-q|, |f-M|] per (b,t).
// 4 waves / block: waves cover 64 consecutive rows (4x16), sharing one staged
// 32x16 W1 tile per K-chunk.
__device__ __forceinline__ float featval(const float* __restrict__ f,
                                         const float* __restrict__ m,
                                         const float* __restrict__ q,
                                         int s, int j) {
  switch (s) {
    case 0:  return f[j];
    case 1:  return m[j];
    case 2:  return q[j];
    case 3:  return f[j] * q[j];
    case 4:  return f[j] * m[j];
    case 5:  return fabsf(f[j] - q[j]);
    default: return fabsf(f[j] - m[j]);
  }
}

__global__ __launch_bounds__(128) void attn_hidden_wmma(
    const float* __restrict__ fact, const float* __restrict__ mem,
    const float* __restrict__ qv,
    const float* __restrict__ W1, const float* __restrict__ b1,
    float* __restrict__ hidden)
{
  __shared__ __align__(16) __bf16 lB[32][24];

  const int tid  = threadIdx.x;
  const int wid  = tid >> 5;
  const int lane = tid & 31;
  const int half = lane >> 4;
  const int idx  = lane & 15;
  const int tn = blockIdx.x * 16;
  const int tm = blockIdx.y * 64 + wid * 16;

  const int lr  = tid >> 2;        // 0..31: k row of W1 tile
  const int lc4 = (tid & 3) * 4;   // 0,4,8,12: n chunk

  const int row = tm + idx;          // row in [0, B*T)
  const int b   = row >> 9;          // row / T
  const float* fr = fact + (long)row * Hc;
  const float* mr = mem  + (long)b * Hc;
  const float* qr = qv   + (long)b * Hc;

  const int kb = half * 8;

  v8f acc = {};
  for (int k0 = 0; k0 < 7 * Hc; k0 += 32) {
    // stage W1 tile (k = k0+lr, n = tn+lc4 .. +3)
    {
      const float4 w4 = *(const float4*)(W1 + (long)(k0 + lr) * Hc + tn + lc4);
      lB[lr][lc4 + 0] = (__bf16)w4.x;
      lB[lr][lc4 + 1] = (__bf16)w4.y;
      lB[lr][lc4 + 2] = (__bf16)w4.z;
      lB[lr][lc4 + 3] = (__bf16)w4.w;
    }
    __syncthreads();

    const int s  = k0 >> 9;                    // feature segment (chunk-uniform)
    const int jb = (k0 & (Hc - 1)) + kb;       // j within segment
    v16bf a, bq;
#pragma unroll
    for (int i = 0; i < 8; ++i) {
      a[i]      = (__bf16)featval(fr, mr, qr, s, jb + i);
      a[i + 8]  = (__bf16)featval(fr, mr, qr, s, jb + 16 + i);
      bq[i]     = lB[kb + i][idx];
      bq[i + 8] = lB[kb + 16 + i][idx];
    }
    acc = __builtin_amdgcn_wmma_f32_16x16x32_bf16(false, a, false, bq,
                                                  (short)0, acc, false, false);
    __syncthreads();
  }

  const int n = tn + idx;
  const float bv = b1[n];
#pragma unroll
  for (int v = 0; v < 8; ++v) {
    const int m = tm + half * 8 + v;
    hidden[(long)m * Hc + n] = tanhf(acc[v] + bv);
  }
}

// ------------------------------- elementwise ---------------------------------
__global__ void fill_f32(float* __restrict__ p, long n, float v) {
  long i = blockIdx.x * (long)blockDim.x + threadIdx.x;
  const long st = (long)gridDim.x * blockDim.x;
  for (; i < n; i += st) p[i] = v;
}

__global__ void copy_f32(float* __restrict__ d, const float* __restrict__ s, long n) {
  long i = blockIdx.x * (long)blockDim.x + threadIdx.x;
  const long st = (long)gridDim.x * blockDim.x;
  for (; i < n; i += st) d[i] = s[i];
}

__global__ void ewise_mul(const float* __restrict__ a, const float* __restrict__ b,
                          float* __restrict__ o, long n) {
  long i = blockIdx.x * (long)blockDim.x + threadIdx.x;
  const long st = (long)gridDim.x * blockDim.x;
  for (; i < n; i += st) o[i] = a[i] * b[i];
}

__global__ void gather_rows(const float* __restrict__ emb, const int* __restrict__ ids,
                            float* __restrict__ out, int H) {
  const int r = blockIdx.x;
  const long src = (long)ids[r] * H;
  const long dst = (long)r * H;
  for (int j = threadIdx.x; j < H; j += blockDim.x) out[dst + j] = emb[src + j];
}

// TF reverse_sequence: reverse first len[b] steps, keep the tail.
__global__ void reverse_seq_k(const float* __restrict__ x, const int* __restrict__ len,
                              float* __restrict__ out) {
  const int t = blockIdx.x, b = blockIdx.y;
  const int L = len[b];
  const int s = (t < L) ? (L - 1 - t) : t;
  const float* xp = x   + ((long)b * Tc + s) * Hc;
  float*       op = out + ((long)b * Tc + t) * Hc;
  for (int j = threadIdx.x; j < Hc; j += blockDim.x) op[j] = xp[j];
}

__global__ void fact_combine_k(const float* __restrict__ fw, const float* __restrict__ bw,
                               const int* __restrict__ len, float* __restrict__ fact) {
  const int t = blockIdx.x, b = blockIdx.y;
  const int L = len[b];
  const int rt = (t < L) ? (L - 1 - t) : t;
  const float* fp = fw + ((long)b * Tc + t) * Hc;
  const float* bp = bw + ((long)b * Tc + rt) * Hc;
  float* op = fact + ((long)b * Tc + t) * Hc;
  for (int j = threadIdx.x; j < Hc; j += blockDim.x) op[j] = fp[j] + bp[j];
}

// GRU: rh = sigmoid(zg[:, :H]) * h
__global__ void gru_gate1(const float* __restrict__ zg, const float* __restrict__ h,
                          float* __restrict__ rh) {
  const int b = blockIdx.x;
  for (int j = threadIdx.x; j < Hc; j += blockDim.x)
    rh[b * Hc + j] = sigmf(zg[(long)b * 2 * Hc + j]) * h[b * Hc + j];
}

// GRU: u = sigmoid(zg[:, H:]); h = mask ? u*h + (1-u)*c : h   (c = tanh already)
__global__ void gru_gate2(const float* __restrict__ zg, const float* __restrict__ cbuf,
                          const int* __restrict__ qlen, int t, float* __restrict__ h) {
  const int b = blockIdx.x;
  const bool m = t < qlen[b];
  for (int j = threadIdx.x; j < Hc; j += blockDim.x) {
    const float u  = sigmf(zg[(long)b * 2 * Hc + Hc + j]);
    const float hv = h[b * Hc + j];
    h[b * Hc + j] = m ? (u * hv + (1.0f - u) * cbuf[b * Hc + j]) : hv;
  }
}

// LSTM gates (TF LSTMCell, forget_bias=1, gate order i,j,f,o), biases already in z.
__global__ void lstm_gate(const float* __restrict__ z, float* __restrict__ h,
                          float* __restrict__ c, float* __restrict__ outs,
                          const int* __restrict__ len, int t) {
  const int b = blockIdx.x;
  const bool m = t < len[b];
  const float* zb = z + (long)b * 4 * Hc;
  for (int j = threadIdx.x; j < Hc; j += blockDim.x) {
    const float iv = zb[j], jv = zb[Hc + j], fv = zb[2 * Hc + j], ov = zb[3 * Hc + j];
    const float cn = sigmf(fv + 1.0f) * c[b * Hc + j] + sigmf(iv) * tanhf(jv);
    const float hn = sigmf(ov) * tanhf(cn);
    const long oi = ((long)b * Tc + t) * Hc + j;
    if (m) { c[b * Hc + j] = cn; h[b * Hc + j] = hn; outs[oi] = hn; }
    else   { outs[oi] = 0.0f; }
  }
}

// AGRU: h = mask ? g*hh + (1-g)*h : h    (hh = tanh already, from GEMM epilogue)
__global__ void agru_up(const float* __restrict__ hh, const float* __restrict__ g,
                        const int* __restrict__ len, int t, float* __restrict__ h) {
  const int b = blockIdx.x;
  const bool m = t < len[b];
  const float gt = g[(long)b * Tc + t];
  for (int j = threadIdx.x; j < Hc; j += blockDim.x) {
    const float hv = h[b * Hc + j];
    h[b * Hc + j] = m ? (gt * hh[b * Hc + j] + (1.0f - gt) * hv) : hv;
  }
}

// att[row] = hidden[row,:] . W2 + b2  (one wave32 per row)
__global__ void dot_w2(const float* __restrict__ hidden, const float* __restrict__ w2,
                       const float* __restrict__ b2, float* __restrict__ att, int rows) {
  const int gid  = blockIdx.x * blockDim.x + threadIdx.x;
  const int row  = gid >> 5;
  const int lane = threadIdx.x & 31;
  if (row >= rows) return;
  const float* hr = hidden + (long)row * Hc;
  float s = 0.0f;
  for (int i = lane; i < Hc; i += 32) s += hr[i] * w2[i];
#pragma unroll
  for (int o = 16; o > 0; o >>= 1) s += __shfl_down(s, o);
  if (lane == 0) att[row] = s + b2[0];
}

// softmax over T per batch row (over all T, as in reference)
__global__ void softmax_rows(const float* __restrict__ att, float* __restrict__ g) {
  __shared__ float red[256];
  const int b = blockIdx.x, tid = threadIdx.x;
  const float* a = att + (long)b * Tc;
  float mx = -3.4e38f;
  for (int t = tid; t < Tc; t += 256) mx = fmaxf(mx, a[t]);
  red[tid] = mx; __syncthreads();
  for (int s = 128; s > 0; s >>= 1) { if (tid < s) red[tid] = fmaxf(red[tid], red[tid + s]); __syncthreads(); }
  mx = red[0]; __syncthreads();
  float sm = 0.0f;
  for (int t = tid; t < Tc; t += 256) sm += __expf(a[t] - mx);
  red[tid] = sm; __syncthreads();
  for (int s = 128; s > 0; s >>= 1) { if (tid < s) red[tid] += red[tid + s]; __syncthreads(); }
  const float inv = 1.0f / red[0];
  for (int t = tid; t < Tc; t += 256) g[(long)b * Tc + t] = __expf(a[t] - mx) * inv;
}

__global__ void argmax_pair(const float* __restrict__ sl, const float* __restrict__ el,
                            int* __restrict__ ps, int* __restrict__ pe) {
  const int b = threadIdx.x;
  if (b >= Bc) return;
  const float* s = sl + (long)b * Tc;
  const float* e = el + (long)b * Tc;
  float bs = s[0], be = e[0];
  int is = 0, ie = 0;
  for (int t = 1; t < Tc; ++t) {
    if (s[t] > bs) { bs = s[t]; is = t; }
    if (e[t] > be) { be = e[t]; ie = t; }
  }
  ps[b] = is; pe[b] = ie;
}

__global__ void concat5_k(const float* __restrict__ p0, const float* __restrict__ p1,
                          const float* __restrict__ p2, const float* __restrict__ p3,
                          const float* __restrict__ p4, float* __restrict__ out) {
  const int b = blockIdx.x;
  const long ob = (long)b * 5 * Hc;
  for (int j = threadIdx.x; j < Hc; j += blockDim.x) {
    out[ob + 0 * Hc + j] = p0[b * Hc + j];
    out[ob + 1 * Hc + j] = p1[b * Hc + j];
    out[ob + 2 * Hc + j] = p2[b * Hc + j];
    out[ob + 3 * Hc + j] = p3[b * Hc + j];
    out[ob + 4 * Hc + j] = p4[b * Hc + j];
  }
}

__global__ void concat2_k(const float* __restrict__ p0, const float* __restrict__ p1,
                          float* __restrict__ out) {
  const int b = blockIdx.x;
  const long ob = (long)b * 2 * Hc;
  for (int j = threadIdx.x; j < Hc; j += blockDim.x) {
    out[ob + j]      = p0[b * Hc + j];
    out[ob + Hc + j] = p1[b * Hc + j];
  }
}

} // namespace

extern "C" void kernel_launch(void* const* d_in, const int* in_sizes, int n_in,
                              void* d_out, int out_size, void* d_ws, size_t ws_size,
                              hipStream_t stream) {
  (void)in_sizes; (void)n_in; (void)out_size; (void)ws_size;

  const int*   question  = (const int*)d_in[0];
  const int*   input_ids = (const int*)d_in[1];
  const int*   qlen      = (const int*)d_in[2];
  const int*   ilen      = (const int*)d_in[3];
  const int*   start     = (const int*)d_in[4];
  const float* emb       = (const float*)d_in[5];
  const float* Wg  = (const float*)d_in[6];
  const float* bg  = (const float*)d_in[7];
  const float* Wc  = (const float*)d_in[8];
  const float* bc  = (const float*)d_in[9];
  const float* lfW = (const float*)d_in[10];
  const float* lfb = (const float*)d_in[11];
  const float* lbW = (const float*)d_in[12];
  const float* lbb = (const float*)d_in[13];
  const float* aW1 = (const float*)d_in[14];
  const float* ab1 = (const float*)d_in[15];
  const float* aW2 = (const float*)d_in[16];
  const float* ab2 = (const float*)d_in[17];
  const float* gWr = (const float*)d_in[18];
  const float* gUr = (const float*)d_in[19];
  const float* gbr = (const float*)d_in[20];
  const float* gWc = (const float*)d_in[21];
  const float* gUc = (const float*)d_in[22];
  const float* gbc = (const float*)d_in[23];
  const float* hopW = (const float*)d_in[24];
  const float* hopb = (const float*)d_in[25];
  const float* osW  = (const float*)d_in[26];
  const float* osb  = (const float*)d_in[27];
  const float* oeW  = (const float*)d_in[28];
  const float* oeb  = (const float*)d_in[29];
  float* out = (float*)d_out;

  // ---- workspace carve (floats); aliasing noted inline ----
  float* w = (float*)d_ws;
  size_t off = 0;
  auto take = [&](size_t n) { float* p = w + off; off += n; return p; };
  float* q_emb  = take((size_t)Bc * TQc * Hc);
  float* x_emb  = take((size_t)Bc * Tc * Hc);
  float* x_rev  = take((size_t)Bc * Tc * Hc);   // -> fact after bw LSTM
  float* fw_out = take((size_t)Bc * Tc * Hc);   // -> attn hidden -> zrf per hop
  float* bw_out = take((size_t)Bc * Tc * Hc);   // -> zcf per hop
  float* attb   = take((size_t)Bc * Tc);
  float* gbuf   = take((size_t)Bc * Tc);
  float* q_vec  = take((size_t)Bc * Hc);
  float* h_st   = take((size_t)Bc * Hc);
  float* c_st   = take((size_t)Bc * Hc);
  float* rh     = take((size_t)Bc * Hc);
  float* z1     = take((size_t)Bc * 4 * Hc);
  float* prev_m = take((size_t)Bc * Hc);
  float* a_s    = take((size_t)Bc * Hc);
  float* a_e    = take((size_t)Bc * Hc);
  float* cat5   = take((size_t)Bc * 5 * Hc);
  float* cat2   = take((size_t)Bc * 2 * Hc);
  float* s_log  = take((size_t)Bc * Tc);
  float* e_log  = take((size_t)Bc * Tc);
  int*   pred_s = (int*)take(Bc);
  int*   pred_e = (int*)take(Bc);

  float* fact   = x_rev;
  float* hidden = fw_out;
  float* zrf    = fw_out;
  float* zcf    = bw_out;

  auto GEMM = [&](const float* A, long lda, const float* Bm, long ldb,
                  const float* Cin, long ldcin, const float* bias,
                  float* C, long ldc, int M, int N, int K, int act) {
    gemm_wmma_bf16<<<dim3((unsigned)(N / 32), (unsigned)(M / 32)), dim3(128), 0, stream>>>(
        A, lda, Bm, ldb, Cin, ldcin, bias, C, ldc, M, N, K, act);
  };

  // ---- embedding lookups ----
  gather_rows<<<Bc * TQc, 256, 0, stream>>>(emb, question, q_emb, Hc);
  gather_rows<<<Bc * Tc, 256, 0, stream>>>(emb, input_ids, x_emb, Hc);

  // ---- question GRU -> q_vec ----
  fill_f32<<<64, 256, 0, stream>>>(q_vec, (long)Bc * Hc, 0.0f);
  for (int t = 0; t < TQc; ++t) {
    const float* xt = q_emb + (long)t * Hc;
    // zg = x_t @ Wg[:H] + bg ; zg += h @ Wg[H:]
    GEMM(xt, (long)TQc * Hc, Wg, 2 * Hc, nullptr, 0, bg, z1, 2 * Hc, Bc, 2 * Hc, Hc, 0);
    GEMM(q_vec, Hc, Wg + (size_t)Hc * 2 * Hc, 2 * Hc, z1, 2 * Hc, nullptr, z1, 2 * Hc, Bc, 2 * Hc, Hc, 0);
    gru_gate1<<<Bc, 256, 0, stream>>>(z1, q_vec, rh);
    float* zc = z1 + (size_t)Bc * 2 * Hc;
    // c = tanh(x_t @ Wc[:H] + bc + (r*h) @ Wc[H:])
    GEMM(xt, (long)TQc * Hc, Wc, Hc, nullptr, 0, bc, zc, Hc, Bc, Hc, Hc, 0);
    GEMM(rh, Hc, Wc + (size_t)Hc * Hc, Hc, zc, Hc, nullptr, zc, Hc, Bc, Hc, Hc, 2);
    gru_gate2<<<Bc, 256, 0, stream>>>(z1, zc, qlen, t, q_vec);
  }

  // ---- bi-LSTM over input ----
  reverse_seq_k<<<dim3(Tc, Bc), 256, 0, stream>>>(x_emb, ilen, x_rev);

  fill_f32<<<64, 256, 0, stream>>>(h_st, (long)Bc * Hc, 0.0f);
  fill_f32<<<64, 256, 0, stream>>>(c_st, (long)Bc * Hc, 0.0f);
  for (int t = 0; t < Tc; ++t) {
    const float* xt = x_emb + (long)t * Hc;
    GEMM(xt, (long)Tc * Hc, lfW, 4 * Hc, nullptr, 0, lfb, z1, 4 * Hc, Bc, 4 * Hc, Hc, 0);
    GEMM(h_st, Hc, lfW + (size_t)Hc * 4 * Hc, 4 * Hc, z1, 4 * Hc, nullptr, z1, 4 * Hc, Bc, 4 * Hc, Hc, 0);
    lstm_gate<<<Bc, 256, 0, stream>>>(z1, h_st, c_st, fw_out, ilen, t);
  }

  fill_f32<<<64, 256, 0, stream>>>(h_st, (long)Bc * Hc, 0.0f);
  fill_f32<<<64, 256, 0, stream>>>(c_st, (long)Bc * Hc, 0.0f);
  for (int t = 0; t < Tc; ++t) {
    const float* xt = x_rev + (long)t * Hc;
    GEMM(xt, (long)Tc * Hc, lbW, 4 * Hc, nullptr, 0, lbb, z1, 4 * Hc, Bc, 4 * Hc, Hc, 0);
    GEMM(h_st, Hc, lbW + (size_t)Hc * 4 * Hc, 4 * Hc, z1, 4 * Hc, nullptr, z1, 4 * Hc, Bc, 4 * Hc, Hc, 0);
    lstm_gate<<<Bc, 256, 0, stream>>>(z1, h_st, c_st, bw_out, ilen, t);
  }

  // fact = fw + reverse(bw)   (writes into x_rev alias; fw/bw then freed)
  fact_combine_k<<<dim3(Tc, Bc), 256, 0, stream>>>(fw_out, bw_out, ilen, fact);

  // ---- episodic memory hops ----
  copy_f32<<<64, 256, 0, stream>>>(prev_m, q_vec, (long)Bc * Hc);
  fill_f32<<<64, 256, 0, stream>>>(a_s, (long)Bc * Hc, 0.0f);
  fill_f32<<<64, 256, 0, stream>>>(a_e, (long)Bc * Hc, 0.0f);

  for (int i = 0; i < 3; ++i) {
    // attention: hidden = tanh(feats @ W1 + b1), att = hidden.W2 + b2, g = softmax_T
    attn_hidden_wmma<<<dim3(Hc / 16, (Bc * Tc) / 64), dim3(128), 0, stream>>>(
        fact, prev_m, q_vec, aW1, ab1, hidden);
    dot_w2<<<(Bc * Tc) / 8, 256, 0, stream>>>(hidden, aW2, ab2, attb, Bc * Tc);
    softmax_rows<<<Bc, 256, 0, stream>>>(attb, gbuf);

    // hoisted AGRU input projections for all T (overwrites hidden)
    GEMM(fact, Hc, gWr, Hc, nullptr, 0, gbr, zrf, Hc, Bc * Tc, Hc, Hc, 0);
    GEMM(fact, Hc, gWc, Hc, nullptr, 0, gbc, zcf, Hc, Bc * Tc, Hc, Hc, 0);

    fill_f32<<<64, 256, 0, stream>>>(h_st, (long)Bc * Hc, 0.0f);
    float* hh = z1 + (size_t)Bc * Hc;
    for (int t = 0; t < Tc; ++t) {
      // r = sigmoid(zrf_t + h @ Ur)   (sigmoid fused into GEMM epilogue)
      GEMM(h_st, Hc, gUr, Hc, zrf + (long)t * Hc, (long)Tc * Hc, nullptr, z1, Hc, Bc, Hc, Hc, 3);
      ewise_mul<<<64, 256, 0, stream>>>(z1, h_st, rh, (long)Bc * Hc);
      // hh = tanh(zcf_t + (r*h) @ Uc)
      GEMM(rh, Hc, gUc, Hc, zcf + (long)t * Hc, (long)Tc * Hc, nullptr, hh, Hc, Bc, Hc, Hc, 2);
      agru_up<<<Bc, 256, 0, stream>>>(hh, gbuf, ilen, t, h_st);
    }
    // episode = h_st

    if (i == 2) {  // last hop: a_s = a_e = emb[start]
      gather_rows<<<Bc, 256, 0, stream>>>(emb, start, a_s, Hc);
      gather_rows<<<Bc, 256, 0, stream>>>(emb, start, a_e, Hc);
    }

    // memory update: prev_m = relu([prev_m, episode, q, a_s, a_e] @ hopW[i] + hopb[i])
    concat5_k<<<Bc, 256, 0, stream>>>(prev_m, h_st, q_vec, a_s, a_e, cat5);
    GEMM(cat5, 5 * Hc, hopW + (size_t)i * 5 * Hc * Hc, Hc, nullptr, 0,
         hopb + (size_t)i * Hc, prev_m, Hc, Bc, Hc, 5 * Hc, 1);

    if (i != 1) {  // hop-1 outputs are dead (neither returned nor used for preds)
      concat2_k<<<Bc, 256, 0, stream>>>(prev_m, q_vec, cat2);
      float* ds = (i == 2) ? out : s_log;
      float* de = (i == 2) ? (out + (size_t)Bc * Tc) : e_log;
      GEMM(cat2, 2 * Hc, osW, Tc, nullptr, 0, osb, ds, Tc, Bc, Tc, 2 * Hc, 0);
      GEMM(cat2, 2 * Hc, oeW, Tc, nullptr, 0, oeb, de, Tc, Bc, Tc, 2 * Hc, 0);
    }

    if (i == 0) {  // i < HOPS-2: pred swap (replicated from reference)
      argmax_pair<<<1, 64, 0, stream>>>(s_log, e_log, pred_s, pred_e);
      gather_rows<<<Bc, 256, 0, stream>>>(emb, pred_s, a_e, Hc);  // a_e = emb[pred_s]
      gather_rows<<<Bc, 256, 0, stream>>>(emb, pred_e, a_s, Hc);  // a_s = emb[pred_e]
    }
  }
}
